// CHAREncoder_33500744909048
// MI455X (gfx1250) — matrix operand order repaired and checked by reading, not compiled, not run on hardware
//
#include <hip/hip_runtime.h>
#include <hip/hip_bf16.h>

// ---- problem constants (match reference) ----
#define Bsz   64
#define Wsz   256
#define Lsz   25
#define Esz   32
#define Hsz   32
#define Nseq  (Bsz * Wsz)     // 16384 sequences
#define NTILE (Nseq / 16)     // 1024 tiles of 16 sequences

typedef __attribute__((ext_vector_type(16))) _Float16 v16h;
typedef __attribute__((ext_vector_type(8)))  _Float16 v8h;
typedef __attribute__((ext_vector_type(8)))  float    v8f;

union V16u { v16h v; v8h h[2]; };

// Hardware tanh (V_TANH_F32 is a gfx1250 TRANS op); exp+fast-rcp fallback.
__device__ __forceinline__ float tanh_hw(float x) {
#if __has_builtin(__builtin_amdgcn_tanhf)
    return __builtin_amdgcn_tanhf(x);
#elif __has_builtin(__builtin_amdgcn_tanh_f32)
    return __builtin_amdgcn_tanh_f32(x);
#else
    // saturates correctly at +/-1; raw v_rcp_f32, no IEEE div fixup
    float e = __expf(2.0f * x);
    return 1.0f - 2.0f * __builtin_amdgcn_rcpf(e + 1.0f);
#endif
}
// sigmoid via tanh: 1 TRANS + mul + fma (avoids IEEE divide sequences)
__device__ __forceinline__ float sigm(float x) {
    return 0.5f * tanh_hw(0.5f * x) + 0.5f;
}

// Two wave32s per (tile of 16 sequences, direction). Wave w owns hidden
// columns 16w..16w+15, i.e. gate tiles {w, 2+w, 4+w, 6+w} (i,f,g,o).
// Per step each wave: 8 x v_wmma_f32_16x16x32_f16 (x-proj + h-proj chained
// into bias-initialized f32 accumulators for its 4 gate tiles), lane-local
// LSTM cell update for its half, h relayout C(f32)->A(f16) via LDS, and a
// barrier pair to exchange h halves between the two waves.
__global__ __launch_bounds__(64)
void char_bilstm_wmma(const int*   __restrict__ ids,   // [Nseq][Lsz] char ids
                      const float* __restrict__ emb,   // [V][E]
                      const float* __restrict__ wihF, const float* __restrict__ whhF,
                      const float* __restrict__ bihF, const float* __restrict__ bhhF,
                      const float* __restrict__ wihB, const float* __restrict__ whhB,
                      const float* __restrict__ bihB, const float* __restrict__ bhhB,
                      float* __restrict__ out)         // [Nseq][2H]
{
    __shared__ __align__(16) _Float16 xbuf[Lsz * 16 * 32]; // [t][m][e] f16
    __shared__ __align__(16) _Float16 hbuf[16 * 32];       // [m][j]    f16

    const int tid  = threadIdx.x;   // 0..63
    const int lane = tid & 31;
    const int w    = tid >> 5;      // wave id == hidden-column half (0 or 1)
    const int tile = blockIdx.x;    // 0..NTILE-1
    const int dir  = blockIdx.y;    // 0 = fwd, 1 = bwd
    const int lrow = lane & 15;
    const int lhi  = lane >> 4;     // 0 or 1

    const float* w_ih = dir ? wihB : wihF;
    const float* w_hh = dir ? whhB : whhF;
    const float* b_ih = dir ? bihB : bihF;
    const float* b_hh = dir ? bhhB : bhhF;

    // ---- Preload this wave's 4 gate tiles of each weight matrix as
    // B-layout f16 (K=32, N=16 per tile). Tile k holds gate tile T = w + 2k
    // (k=0:i, 1:f, 2:g, 3:o). B-layout: lane l holds column n = l%16
    // (== weight row 16T + l%16), elements 0..7 = K 8*(l/16)+0..7, 8..15 = K 16+8*(l/16)+0..7.
    v16h wih[4], whh[4];
    float bias[4];
    #pragma unroll
    for (int k = 0; k < 4; ++k) {
        const int row = 16 * (w + 2 * k) + lrow;
        const int kb  = lhi * 8;
        const float* pi = w_ih + row * Esz + kb;
        const float* ph = w_hh + row * Hsz + kb;
        V16u a, b;
        #pragma unroll
        for (int j = 0; j < 16; ++j) {
            a.v[j] = (_Float16)pi[j + (j & 8)];
            b.v[j] = (_Float16)ph[j + (j & 8)];
        }
        wih[k] = a.v;
        whh[k] = b.v;
        bias[k] = b_ih[row] + b_hh[row];
    }

    // ---- Gather embeddings for all 25 steps x 16 rows into LDS (f16).
    for (int p = tid; p < Lsz * 16; p += 64) {
        const int t = p >> 4;
        const int m = p & 15;
        const int id = ids[(tile * 16 + m) * Lsz + t];
        const float* er = emb + id * Esz;
        _Float16* dst = &xbuf[(t * 16 + m) * 32];
        #pragma unroll
        for (int j = 0; j < 32; j += 8) {
            v8h v;
            #pragma unroll
            for (int kk = 0; kk < 8; ++kk) v[kk] = (_Float16)er[j + kk];
            *(v8h*)(dst + j) = v;
        }
    }
    // zero h buffer (512 halfs, 8 per thread)
    {
        v8h z = {};
        *(v8h*)(&hbuf[tid * 8]) = z;
    }
    __syncthreads();

    // A-layout (16-bit 16x32): lane l holds row l%16;
    // elems 0..7 = K 8*(l/16)+0..7, elems 8..15 = K 16+8*(l/16)+0..7
    const int abase = lrow * 32 + lhi * 8;

    v8f c    = {};   // cell state for this wave's 16 hidden cols (C layout)
    v8f hout = {};   // last hidden (f32) for output

    for (int s = 0; s < Lsz; ++s) {
        const int t = dir ? (Lsz - 1 - s) : s;

        V16u xa, ha;
        xa.h[0] = *(const v8h*)&xbuf[t * 512 + abase];
        xa.h[1] = *(const v8h*)&xbuf[t * 512 + abase + 16];
        ha.h[0] = *(const v8h*)&hbuf[abase];
        ha.h[1] = *(const v8h*)&hbuf[abase + 16];

        // 4 gate tiles for this half: k=0:i, 1:f, 2:g, 3:o
        v8f g[4];
        #pragma unroll
        for (int k = 0; k < 4; ++k) {
            v8f acc;
            #pragma unroll
            for (int r = 0; r < 8; ++r) acc[r] = bias[k];
            acc = __builtin_amdgcn_wmma_f32_16x16x32_f16(
                    false, xa.v, false, wih[k], (short)0, acc, false, false);
            acc = __builtin_amdgcn_wmma_f32_16x16x32_f16(
                    false, ha.v, false, whh[k], (short)0, acc, false, false);
            g[k] = acc;
        }

        __syncthreads();  // both waves finished reading hbuf

        // Lane-local LSTM cell update; write h half back to LDS as f16.
        #pragma unroll
        for (int r = 0; r < 8; ++r) {
            const float ig = sigm   (g[0][r]);
            const float fg = sigm   (g[1][r]);
            const float gc = tanh_hw(g[2][r]);
            const float og = sigm   (g[3][r]);
            const float cv = fg * c[r] + ig * gc;
            c[r] = cv;
            const float hv = og * tanh_hw(cv);
            hout[r] = hv;
            // C layout: row m = r + 8*(l/16), col j = 16*w + l%16
            hbuf[(r + 8 * lhi) * 32 + 16 * w + lrow] = (_Float16)hv;
        }

        __syncthreads();  // h halves visible to both waves
    }

    // ---- final hidden -> out[n][dir*32 + j], f32
    #pragma unroll
    for (int r = 0; r < 8; ++r) {
        const int n = tile * 16 + r + 8 * lhi;
        out[n * (2 * Hsz) + dir * Hsz + 16 * w + lrow] = hout[r];
    }
}

extern "C" void kernel_launch(void* const* d_in, const int* in_sizes, int n_in,
                              void* d_out, int out_size, void* d_ws, size_t ws_size,
                              hipStream_t stream) {
    (void)in_sizes; (void)n_in; (void)d_ws; (void)ws_size; (void)out_size;
    dim3 grid(NTILE, 2);   // 1024 sequence tiles x 2 directions
    char_bilstm_wmma<<<grid, 64, 0, stream>>>(
        (const int*)  d_in[0],   // char_ids
        (const float*)d_in[1],   // emb
        (const float*)d_in[2], (const float*)d_in[3],   // w_ih_f, w_hh_f
        (const float*)d_in[4], (const float*)d_in[5],   // b_ih_f, b_hh_f
        (const float*)d_in[6], (const float*)d_in[7],   // w_ih_b, w_hh_b
        (const float*)d_in[8], (const float*)d_in[9],   // b_ih_b, b_hh_b
        (float*)d_out);
}
